// DynamicK_57964878627029
// MI455X (gfx1250) — matrix operand order repaired and checked by reading, not compiled, not run on hardware
//
#include <hip/hip_runtime.h>
#include <hip/hip_bf16.h>

#define NTOK   16384
#define DMODEL 2048
#define NEXP   64
#define THRESH 0.9f

typedef __bf16 v16bf __attribute__((ext_vector_type(16)));
typedef float  v8f   __attribute__((ext_vector_type(8)));
typedef float  v4f   __attribute__((ext_vector_type(4)));

// ---------------------------------------------------------------------------
// Kernel 1: prepack W [2048 x 64] f32 row-major into bf16 WMMA-B lane layout.
// B k order used here (contiguous; A matches): lanes 0-15 hold N=lane, K = kc*32 + 0..15;
// lanes 16-31 hold N=lane-16, K = kc*32 + 16..31; element i = K base+i.
// ---------------------------------------------------------------------------
__global__ void __launch_bounds__(256)
prepack_B_kernel(const float* __restrict__ W, v16bf* __restrict__ wsB) {
    int tid  = blockIdx.x * blockDim.x + threadIdx.x;   // 8192 threads total
    int lane = tid & 31;
    int nt   = (tid >> 5) & 3;
    int kc   = tid >> 7;                                // 0..63
    int n     = nt * 16 + (lane & 15);
    int kbase = kc * 32 + ((lane >> 4) ? 16 : 0);
    v16bf v;
#pragma unroll
    for (int i = 0; i < 16; ++i)
        v[i] = (__bf16)W[(size_t)(kbase + i) * NEXP + n];
    wsB[tid] = v;
}

// ---------------------------------------------------------------------------
// Kernel 2: fused GEMM (bf16 WMMA, f32 accum) + bias + softmax + order-free
// nucleus selection. One wave = 16 tokens x 64 experts; LDS tile wave-private.
// K loop: software pipeline unrolled x2 with two named buffer sets so the
// steady state has zero register copies and next-chunk loads are always in
// flight across the current chunk's WMMAs.
// ---------------------------------------------------------------------------
__global__ void __launch_bounds__(128)
moe_route_kernel(const float* __restrict__ x, const v16bf* __restrict__ wsB,
                 const float* __restrict__ bias, float* __restrict__ out) {
    __shared__ float lds[4][16][68];                    // 68 = 17*4: 16B-aligned rows

    const int lane    = threadIdx.x & 31;
    const int wave    = threadIdx.x >> 5;
    const int tokBase = (blockIdx.x * 4 + wave) * 16;
    const int mrow    = lane & 15;                      // token row within tile
    const int grp     = lane >> 4;
    const float* xr   = x + (size_t)(tokBase + mrow) * DMODEL + grp * 16;

    v8f acc[4];
#pragma unroll
    for (int nt = 0; nt < 4; ++nt)
        acc[nt] = (v8f){0.f, 0.f, 0.f, 0.f, 0.f, 0.f, 0.f, 0.f};

    // A k order used here (contiguous, same as B): lane grp 0 holds K {c..c+15},
    // grp 1 holds K {c+16..c+31}; lanes L and L+16 together cover
    // one full 128B cacheline of row L per chunk. x is streamed once -> NT.
    auto loadB = [&](int kc, v16bf& b0, v16bf& b1, v16bf& b2, v16bf& b3) {
        const v16bf* bp = wsB + (size_t)kc * 128 + lane;
        b0 = bp[0]; b1 = bp[32]; b2 = bp[64]; b3 = bp[96];
    };
    auto loadA = [&](int kc, v4f& f0, v4f& f1, v4f& f2, v4f& f3) {
        const float* ap = xr + kc * 32;
        f0 = __builtin_nontemporal_load((const v4f*)(ap));
        f1 = __builtin_nontemporal_load((const v4f*)(ap + 4));
        f2 = __builtin_nontemporal_load((const v4f*)(ap + 8));
        f3 = __builtin_nontemporal_load((const v4f*)(ap + 12));
        __builtin_prefetch(ap + 8 * 32, 0, 1);          // speculative, OOB dropped
    };
    auto mma = [&](const v4f& f0, const v4f& f1, const v4f& f2, const v4f& f3,
                   const v16bf& b0, const v16bf& b1, const v16bf& b2, const v16bf& b3) {
        v16bf a;
#pragma unroll
        for (int i = 0; i < 4; ++i) {
            a[i]      = (__bf16)f0[i];
            a[4 + i]  = (__bf16)f1[i];
            a[8 + i]  = (__bf16)f2[i];
            a[12 + i] = (__bf16)f3[i];
        }
        acc[0] = __builtin_amdgcn_wmma_f32_16x16x32_bf16(false, a, false, b0,
                                                         (short)0, acc[0], false, false);
        acc[1] = __builtin_amdgcn_wmma_f32_16x16x32_bf16(false, a, false, b1,
                                                         (short)0, acc[1], false, false);
        acc[2] = __builtin_amdgcn_wmma_f32_16x16x32_bf16(false, a, false, b2,
                                                         (short)0, acc[2], false, false);
        acc[3] = __builtin_amdgcn_wmma_f32_16x16x32_bf16(false, a, false, b3,
                                                         (short)0, acc[3], false, false);
    };

    // ---- pipelined K loop: 64 chunks, ping-pong buffer sets ---------------
    v16bf pb0, pb1, pb2, pb3, qb0, qb1, qb2, qb3;       // B sets P and Q
    v4f   pf0, pf1, pf2, pf3, qf0, qf1, qf2, qf3;       // A sets P and Q

    loadB(0, pb0, pb1, pb2, pb3);
    loadA(0, pf0, pf1, pf2, pf3);
#pragma unroll 1
    for (int kc = 0; kc < DMODEL / 32 - 2; kc += 2) {
        loadB(kc + 1, qb0, qb1, qb2, qb3);              // issue Q loads
        loadA(kc + 1, qf0, qf1, qf2, qf3);
        mma(pf0, pf1, pf2, pf3, pb0, pb1, pb2, pb3);    // consume P (kc)

        loadB(kc + 2, pb0, pb1, pb2, pb3);              // issue P loads
        loadA(kc + 2, pf0, pf1, pf2, pf3);
        mma(qf0, qf1, qf2, qf3, qb0, qb1, qb2, qb3);    // consume Q (kc+1)
    }
    loadB(DMODEL / 32 - 1, qb0, qb1, qb2, qb3);         // kc = 63
    loadA(DMODEL / 32 - 1, qf0, qf1, qf2, qf3);
    mma(pf0, pf1, pf2, pf3, pb0, pb1, pb2, pb3);        // kc = 62
    mma(qf0, qf1, qf2, qf3, qb0, qb1, qb2, qb3);        // kc = 63

    // ---- Epilogue: bias + park logits in wave-private LDS tile ------------
    // C layout: lane L holds (M = grp*8 + j, N = L&15) in acc element j.
#pragma unroll
    for (int nt = 0; nt < 4; ++nt) {
        float bb = bias[nt * 16 + mrow];
#pragma unroll
        for (int j = 0; j < 8; ++j)
            lds[wave][grp * 8 + j][nt * 16 + mrow] = acc[nt][j] + bb;
    }

    float* rw    = out;
    float* probs = out + (size_t)NTOK * NEXP;
    float* cnt   = out + (size_t)2 * NTOK * NEXP;

    // ---- Softmax + order-free nucleus selection, 1 token at a time --------
    // Expert e is active iff S_e = sum_{p_j > p_e, or p_j==p_e && j<e} p_j
    // is < 0.9 (== shifted cumsum of the stable descending sort at e).
#pragma unroll 1
    for (int t = 0; t < 16; ++t) {
        const int tok = tokBase + t;
        const int e1  = lane;           // this lane's two experts
        const int e2  = lane + 32;
        float l1 = lds[wave][t][e1];
        float l2 = lds[wave][t][e2];

        float m = fmaxf(l1, l2);
#pragma unroll
        for (int off = 16; off; off >>= 1)
            m = fmaxf(m, __shfl_xor(m, off, 32));

        float x1 = __expf(l1 - m), x2 = __expf(l2 - m);
        float s = x1 + x2;
#pragma unroll
        for (int off = 16; off; off >>= 1)
            s += __shfl_xor(s, off, 32);
        float inv = 1.0f / s;
        float p1 = x1 * inv, p2 = x2 * inv;

        // Overwrite logits with probs, then scan via broadcast quad loads:
        // one ds_load_b128 feeds 4 expert probs to all 32 lanes.
        lds[wave][t][e1] = p1;
        lds[wave][t][e2] = p2;

        float S1 = 0.f, S2 = 0.f;
#pragma unroll 4
        for (int q = 0; q < 16; ++q) {
            v4f pq = *(const v4f*)&lds[wave][t][q * 4];
#pragma unroll
            for (int r = 0; r < 4; ++r) {
                const int j  = q * 4 + r;
                const float pj = pq[r];
                if (pj > p1 || (pj == p1 && j < e1)) S1 += pj;
                if (pj > p2 || (pj == p2 && j < e2)) S2 += pj;
            }
        }
        int a1 = S1 < THRESH;
        int a2 = S2 < THRESH;

        float T = (a1 ? p1 : 0.f) + (a2 ? p2 : 0.f);
#pragma unroll
        for (int off = 16; off; off >>= 1)
            T += __shfl_xor(T, off, 32);
        unsigned c = __popc(__builtin_amdgcn_ballot_w32(a1 != 0)) +
                     __popc(__builtin_amdgcn_ballot_w32(a2 != 0));
        float invT = 1.0f / (T + 1e-6f);

        __builtin_nontemporal_store(p1, &probs[(size_t)tok * NEXP + e1]);
        __builtin_nontemporal_store(p2, &probs[(size_t)tok * NEXP + e2]);
        __builtin_nontemporal_store(a1 ? p1 * invT : 0.f, &rw[(size_t)tok * NEXP + e1]);
        __builtin_nontemporal_store(a2 ? p2 * invT : 0.f, &rw[(size_t)tok * NEXP + e2]);
        if (lane == 0) __builtin_nontemporal_store((float)c, &cnt[tok]);
    }
}

// ---------------------------------------------------------------------------
extern "C" void kernel_launch(void* const* d_in, const int* in_sizes, int n_in,
                              void* d_out, int out_size, void* d_ws, size_t ws_size,
                              hipStream_t stream) {
    const float* x = (const float*)d_in[0];   // [16384, 2048]
    const float* W = (const float*)d_in[1];   // [2048, 64]
    const float* b = (const float*)d_in[2];   // [64]
    float* out     = (float*)d_out;
    v16bf* wsB     = (v16bf*)d_ws;            // 256 KB prepacked bf16 B

    // 64 kchunks * 4 ntiles * 32 lanes = 8192 threads
    prepack_B_kernel<<<32, 256, 0, stream>>>(W, wsB);
    // 16384 tokens / (4 waves * 16 tokens) = 256 blocks
    moe_route_kernel<<<NTOK / 64, 128, 0, stream>>>(x, wsB, b, out);
}
